// DKVMN_29961691857545
// MI455X (gfx1250) — matrix operand order repaired and checked by reading, not compiled, not run on hardware
//
#include <hip/hip_runtime.h>
#include <hip/hip_bf16.h>
#include <math.h>

typedef __attribute__((ext_vector_type(16))) _Float16 v16h;
typedef __attribute__((ext_vector_type(8)))  float    v8f;
typedef __attribute__((__vector_size__(4 * sizeof(int)))) int i4v;

#define B_   128
#define S_   200
#define NT   25600   // B_*S_
#define M_   50
#define DK   64
#define DV   128
#define F_   64
#define KCAT 192     // DV + DK

#define __AS1 __attribute__((address_space(1)))
#define __AS3 __attribute__((address_space(3)))

#if defined(__has_builtin)
#if __has_builtin(__builtin_amdgcn_global_load_async_to_lds_b128) && \
    __has_builtin(__builtin_amdgcn_s_wait_asynccnt)
#define USE_ASYNC_LDS 1
#endif
#endif

// ---------------------------------------------------------------------------
// 1. Embedding gather -> f16 (WMMA A operands)
// ---------------------------------------------------------------------------
__global__ void k_embed(const int* __restrict__ qd, const int* __restrict__ qad,
                        const float* __restrict__ qw, const float* __restrict__ qaw,
                        _Float16* __restrict__ qeh, _Float16* __restrict__ qah) {
    int tid = blockIdx.x * blockDim.x + threadIdx.x;   // NT*128 threads
    int i = tid >> 7, k = tid & 127;
    int qa = qad[i];
    qah[(size_t)i * DV + k] = (_Float16)qaw[(size_t)qa * DV + k];
    if (k < DK) {
        int q = qd[i];
        qeh[(size_t)i * DK + k] = (_Float16)qw[(size_t)q * DK + k];
    }
}

// ---------------------------------------------------------------------------
// 2. Weights -> f16 copies (WMMA B operands); mem_key zero-padded to 64 rows
// ---------------------------------------------------------------------------
__global__ void k_wconv(const float* __restrict__ mem_key, const float* __restrict__ erase_w,
                        const float* __restrict__ add_w, const float* __restrict__ read_w,
                        _Float16* __restrict__ mkh, _Float16* __restrict__ ewh,
                        _Float16* __restrict__ awh, _Float16* __restrict__ rwh) {
    int tid = blockIdx.x * blockDim.x + threadIdx.x;   // 16384 threads
    if (tid < 64 * 64) {
        int n = tid >> 6, k = tid & 63;
        mkh[tid] = (_Float16)((n < M_) ? mem_key[n * DK + k] : 0.0f);
    }
    if (tid < 128 * 128) {
        ewh[tid] = (_Float16)erase_w[tid];
        awh[tid] = (_Float16)add_w[tid];
    }
    if (tid < 64 * 192) {
        rwh[tid] = (_Float16)read_w[tid];
    }
}

// ---------------------------------------------------------------------------
// 3. Fused: e = sigmoid(qa@We^T+be), a = tanh(qa@Wa^T+ba)   [WMMA]
//    One wave owns one 16x16 tile of BOTH outputs (shared A fragment).
// ---------------------------------------------------------------------------
__global__ void k_gemm_ea(const _Float16* __restrict__ qah,
                          const _Float16* __restrict__ ewh, const _Float16* __restrict__ awh,
                          const float* __restrict__ eb, const float* __restrict__ ab,
                          float* __restrict__ eo, float* __restrict__ ao) {
    const int wv   = threadIdx.x >> 5;
    const int lane = threadIdx.x & 31;
    const int hi   = lane >> 4;
    const int l16  = lane & 15;
    const int tile = blockIdx.x * 8 + wv;   // 12800 tiles
    const int ct   = tile & 7;
    const int rt   = tile >> 3;
    const int n    = ct * 16 + l16;
    const float biasE = eb[n];
    const float biasA = ab[n];

    v8f ce = {};
    v8f ca = {};
    #pragma unroll
    for (int kk = 0; kk < 4; ++kk) {
        const int k0 = kk * 32;
        v16h af, bfe, bfa;
        const _Float16* arow = qah + (size_t)(rt * 16 + l16) * DV + k0 + hi * 16;
        #pragma unroll
        for (int e = 0; e < 16; ++e) af[e] = arow[e];
        const _Float16* bce = ewh + (size_t)n * DV + k0 + hi * 16;
        const _Float16* bca = awh + (size_t)n * DV + k0 + hi * 16;
        #pragma unroll
        for (int e = 0; e < 16; ++e) { bfe[e] = bce[e]; bfa[e] = bca[e]; }
        ce = __builtin_amdgcn_wmma_f32_16x16x32_f16(false, af, false, bfe, (short)0, ce, false, false);
        ca = __builtin_amdgcn_wmma_f32_16x16x32_f16(false, af, false, bfa, (short)0, ca, false, false);
    }
    #pragma unroll
    for (int r = 0; r < 8; ++r) {
        int m = rt * 16 + r + hi * 8;
        float ve = ce[r] + biasE;
        float va = ca[r] + biasA;
        eo[(size_t)m * DV + n] = 1.0f / (1.0f + expf(-ve));
        ao[(size_t)m * DV + n] = tanhf(va);
    }
}

// ---------------------------------------------------------------------------
// 4. corr = q_e @ mem_key^T   [25600,64]x[64,64-padded]  [WMMA]
// ---------------------------------------------------------------------------
__global__ void k_gemm_addr(const _Float16* __restrict__ qeh,
                            const _Float16* __restrict__ mkh,
                            float* __restrict__ wbuf) {
    const int wv   = threadIdx.x >> 5;
    const int lane = threadIdx.x & 31;
    const int hi   = lane >> 4;
    const int l16  = lane & 15;
    const int tile = blockIdx.x * 8 + wv;   // 6400 tiles
    const int ct   = tile & 3;
    const int rt   = tile >> 2;
    const int n    = ct * 16 + l16;

    v8f c = {};
    #pragma unroll
    for (int kk = 0; kk < 2; ++kk) {
        const int k0 = kk * 32;
        v16h af, bf;
        const _Float16* arow = qeh + (size_t)(rt * 16 + l16) * DK + k0 + hi * 16;
        #pragma unroll
        for (int e = 0; e < 16; ++e) af[e] = arow[e];
        const _Float16* bcol = mkh + (size_t)n * DK + k0 + hi * 16;
        #pragma unroll
        for (int e = 0; e < 16; ++e) bf[e] = bcol[e];
        c = __builtin_amdgcn_wmma_f32_16x16x32_f16(false, af, false, bf, (short)0, c, false, false);
    }
    #pragma unroll
    for (int r = 0; r < 8; ++r) {
        int m = rt * 16 + r + hi * 8;
        wbuf[(size_t)m * 64 + n] = c[r];
    }
}

// ---------------------------------------------------------------------------
// 5. In-place row softmax over first 50 of each 64-wide row
// ---------------------------------------------------------------------------
__global__ void k_softmax(float* __restrict__ wbuf) {
    int i = blockIdx.x * blockDim.x + threadIdx.x;    // NT threads
    float* row = wbuf + (size_t)i * 64;
    float mx = -1e30f;
    for (int m = 0; m < M_; ++m) mx = fmaxf(mx, row[m]);
    float s = 0.0f;
    for (int m = 0; m < M_; ++m) { float t = expf(row[m] - mx); row[m] = t; s += t; }
    float inv = 1.0f / s;
    for (int m = 0; m < M_; ++m) row[m] *= inv;
}

// ---------------------------------------------------------------------------
// 6. Sequential scan: one workgroup per batch row.
//    All 200 softmax rows (51.2KB) + Mv (25.6KB) staged in LDS via async
//    global->LDS copies; barrier-free main loop (sMv columns are
//    thread-private, sW is read-only); e/a double-buffered in registers.
// ---------------------------------------------------------------------------
__global__ void k_scan(const float* __restrict__ init_mv,
                       const float* __restrict__ wbuf,
                       const float* __restrict__ ebuf,
                       const float* __restrict__ abuf,
                       _Float16* __restrict__ rbh) {
    __shared__ float sMv[M_ * DV];     // 25.6 KB
    __shared__ float sW[S_ * 64];      // 51.2 KB (CDNA5: 320KB LDS/WGP)
    const int b = blockIdx.x;
    const int v = threadIdx.x;         // 128 threads = DV column index
    const float* gW = wbuf + (size_t)b * S_ * 64;

#ifdef USE_ASYNC_LDS
    for (int idx = v * 4; idx < S_ * 64; idx += 128 * 4)
        __builtin_amdgcn_global_load_async_to_lds_b128(
            (__AS1 i4v*)(gW + idx), (__AS3 i4v*)(sW + idx), 0, 0);
    for (int idx = v * 4; idx < M_ * DV; idx += 128 * 4)
        __builtin_amdgcn_global_load_async_to_lds_b128(
            (__AS1 i4v*)(init_mv + idx), (__AS3 i4v*)(sMv + idx), 0, 0);
    __builtin_amdgcn_s_wait_asynccnt(0);
#else
    for (int idx = v; idx < S_ * 64; idx += 128) sW[idx] = gW[idx];
    for (int idx = v; idx < M_ * DV; idx += 128) sMv[idx] = init_mv[idx];
#endif
    __syncthreads();

    const size_t ibase = (size_t)b * S_;
    float ev = ebuf[ibase * DV + v];
    float av = abuf[ibase * DV + v];
    for (int t = 0; t < S_; ++t) {
        float evn = 0.0f, avn = 0.0f;
        if (t + 1 < S_) {                        // register double-buffer
            evn = ebuf[(ibase + t + 1) * DV + v];
            avn = abuf[(ibase + t + 1) * DV + v];
        }
        if (t + 2 < S_) {                        // L2/WGP$ prefetch hint
            __builtin_prefetch(&ebuf[(ibase + t + 2) * DV + v], 0, 1);
            __builtin_prefetch(&abuf[(ibase + t + 2) * DV + v], 0, 1);
        }
        const float* wrow = sW + t * 64;
        float acc = 0.0f;
        #pragma unroll 10
        for (int m = 0; m < M_; ++m) {
            const float wm = wrow[m];
            const float mv = sMv[m * DV + v];
            acc += wm * mv;                                    // read (pre-update)
            sMv[m * DV + v] = mv * (1.0f - wm * ev) + wm * av; // rank-1 update
        }
        rbh[(ibase + t) * DV + v] = (_Float16)acc;
        ev = evn; av = avn;
    }
}

// ---------------------------------------------------------------------------
// 7. h = tanh([read | q_e] @ read_w^T + rb)  [25600,192]x[192,64]  [WMMA]
// ---------------------------------------------------------------------------
__global__ void k_gemm_fc(const _Float16* __restrict__ rbh,
                          const _Float16* __restrict__ qeh,
                          const _Float16* __restrict__ rwh,
                          const float* __restrict__ rb,
                          float* __restrict__ hbuf) {
    const int wv   = threadIdx.x >> 5;
    const int lane = threadIdx.x & 31;
    const int hi   = lane >> 4;
    const int l16  = lane & 15;
    const int tile = blockIdx.x * 8 + wv;   // 6400 tiles
    const int ct   = tile & 3;
    const int rt   = tile >> 2;
    const int n    = ct * 16 + l16;
    const float bias = rb[n];
    const int row  = rt * 16 + l16;

    v8f c = {};
    #pragma unroll
    for (int kk = 0; kk < 6; ++kk) {
        const int k0 = kk * 32 + hi * 16;
        v16h af, bf;
        #pragma unroll
        for (int e = 0; e < 16; ++e) {
            int kg = k0 + e;
            af[e] = (kg < DV) ? rbh[(size_t)row * DV + kg]
                              : qeh[(size_t)row * DK + (kg - DV)];
        }
        const _Float16* bcol = rwh + (size_t)n * KCAT + kk * 32 + hi * 16;
        #pragma unroll
        for (int e = 0; e < 16; ++e) bf[e] = bcol[e];
        c = __builtin_amdgcn_wmma_f32_16x16x32_f16(false, af, false, bf, (short)0, c, false, false);
    }
    #pragma unroll
    for (int r = 0; r < 8; ++r) {
        int m = rt * 16 + r + hi * 8;
        hbuf[(size_t)m * F_ + n] = tanhf(c[r] + bias);
    }
}

// ---------------------------------------------------------------------------
// 8. logits, masked BCE partial sums, probs
// ---------------------------------------------------------------------------
__global__ void k_loss(const float* __restrict__ hbuf,
                       const float* __restrict__ pw, const float* __restrict__ pb,
                       const float* __restrict__ target,
                       float* __restrict__ probs, float* __restrict__ accum) {
    __shared__ float sb[256];
    __shared__ float sc[256];
    const int tid = threadIdx.x;
    const int i = blockIdx.x * 256 + tid;          // NT = 100*256 exactly
    float l = pb[0];
    #pragma unroll 8
    for (int f = 0; f < F_; ++f) l += hbuf[(size_t)i * F_ + f] * pw[f];
    const float t = target[i];
    const bool msk = (t >= 0.0f);
    const float tt = msk ? t : 0.0f;
    const float bce = fmaxf(l, 0.0f) - l * tt + log1pf(expf(-fabsf(l)));
    probs[i] = msk ? (1.0f / (1.0f + expf(-l))) : 0.0f;
    sb[tid] = msk ? bce : 0.0f;
    sc[tid] = msk ? 1.0f : 0.0f;
    __syncthreads();
    for (int st = 128; st > 0; st >>= 1) {
        if (tid < st) { sb[tid] += sb[tid + st]; sc[tid] += sc[tid + st]; }
        __syncthreads();
    }
    if (tid == 0) { atomicAdd(&accum[0], sb[0]); atomicAdd(&accum[1], sc[0]); }
}

__global__ void k_final(const float* __restrict__ accum, float* __restrict__ loss) {
    loss[0] = accum[0] / fmaxf(accum[1], 1.0f);
}

// ---------------------------------------------------------------------------
extern "C" void kernel_launch(void* const* d_in, const int* in_sizes, int n_in,
                              void* d_out, int out_size, void* d_ws, size_t ws_size,
                              hipStream_t stream) {
    const int*   q_data   = (const int*)  d_in[0];
    const int*   qa_data  = (const int*)  d_in[1];
    const float* target   = (const float*)d_in[2];
    const float* q_emb    = (const float*)d_in[3];
    const float* qa_emb   = (const float*)d_in[4];
    const float* mem_key  = (const float*)d_in[5];
    const float* init_mv  = (const float*)d_in[6];
    const float* erase_w  = (const float*)d_in[7];
    const float* erase_b  = (const float*)d_in[8];
    const float* add_w    = (const float*)d_in[9];
    const float* add_b    = (const float*)d_in[10];
    const float* read_w   = (const float*)d_in[11];
    const float* read_b   = (const float*)d_in[12];
    const float* pred_w   = (const float*)d_in[13];
    const float* pred_b   = (const float*)d_in[14];

    float* out = (float*)d_out;          // [0]=loss, [1..]=probs (B*S)

    // Workspace carve-up (256B aligned)
    char* ws = (char*)d_ws;
    size_t o = 0;
    auto take = [&](size_t bytes) { size_t r = o; o += (bytes + 255) & ~(size_t)255; return r; };
    _Float16* qeh  = (_Float16*)(ws + take((size_t)NT * DK * 2));
    _Float16* qah  = (_Float16*)(ws + take((size_t)NT * DV * 2));
    float*    wbuf = (float*)   (ws + take((size_t)NT * 64 * 4));
    float*    ebuf = (float*)   (ws + take((size_t)NT * DV * 4));
    float*    abuf = (float*)   (ws + take((size_t)NT * DV * 4));
    _Float16* rbh  = (_Float16*)(ws + take((size_t)NT * DV * 2));
    float*    hbuf = (float*)   (ws + take((size_t)NT * F_ * 4));
    _Float16* mkh  = (_Float16*)(ws + take(64 * 64 * 2));
    _Float16* ewh  = (_Float16*)(ws + take(128 * 128 * 2));
    _Float16* awh  = (_Float16*)(ws + take(128 * 128 * 2));
    _Float16* rwh  = (_Float16*)(ws + take(64 * 192 * 2));
    float*    accum= (float*)   (ws + take(2 * 4));

    (void)hipMemsetAsync(accum, 0, 2 * sizeof(float), stream);

    k_embed    <<<(NT * 128) / 256, 256, 0, stream>>>(q_data, qa_data, q_emb, qa_emb, qeh, qah);
    k_wconv    <<<64, 256, 0, stream>>>(mem_key, erase_w, add_w, read_w, mkh, ewh, awh, rwh);
    k_gemm_ea  <<<1600, 256, 0, stream>>>(qah, ewh, awh, erase_b, add_b, ebuf, abuf);
    k_gemm_addr<<<800, 256, 0, stream>>>(qeh, mkh, wbuf);
    k_softmax  <<<NT / 256, 256, 0, stream>>>(wbuf);
    k_scan     <<<B_, 128, 0, stream>>>(init_mv, wbuf, ebuf, abuf, rbh);
    k_gemm_fc  <<<800, 256, 0, stream>>>(rbh, qeh, rwh, read_b, hbuf);
    k_loss     <<<NT / 256, 256, 0, stream>>>(hbuf, pred_w, pred_b, target, out + 1, accum);
    k_final    <<<1, 1, 0, stream>>>(accum, out);
}
